// ClusterLookup_3556232921213
// MI455X (gfx1250) — matrix operand order repaired and checked by reading, not compiled, not run on hardware
//
#include <hip/hip_runtime.h>
#include <hip/hip_bf16.h>
#include <math.h>

// Problem constants (from reference): B=32, C=384, H=W=80, N=27, top-k=5
#define C_DIM   384
#define N_CLS   27
#define B_IMG   32
#define HW_IMG  6400
#define N_PIX   (B_IMG * HW_IMG)      // 204800 pixels
#define NBLK    1600                  // 204800 / 128 pixels per block
#define GCALPHA 3.0f

typedef _Float16 v16h __attribute__((ext_vector_type(16)));
typedef float    v8f  __attribute__((ext_vector_type(8)));

// ---------------------------------------------------------------------------
// Kernel 1a: a = tanh(3*clusters@w1), b = tanh(3*clusters@w2), cg = clusters@gcn_w
// One thread per output element; consecutive threads walk consecutive d ->
// coalesced weight reads (w[c*384 + d]).
// ---------------------------------------------------------------------------
__global__ __launch_bounds__(256)
void refine_mm(const float* __restrict__ clusters,
               const float* __restrict__ w1, const float* __restrict__ w2,
               const float* __restrict__ wg,
               float* __restrict__ a, float* __restrict__ b, float* __restrict__ cg)
{
    int t = blockIdx.x * 256 + threadIdx.x;
    if (t >= 3 * N_CLS * C_DIM) return;
    int mat = t / (N_CLS * C_DIM);
    int rem = t % (N_CLS * C_DIM);
    int n = rem / C_DIM, d = rem % C_DIM;
    const float* W    = (mat == 0) ? w1 : ((mat == 1) ? w2 : wg);
    const float* crow = clusters + n * C_DIM;
    float s = 0.f;
    for (int c = 0; c < C_DIM; ++c) s = fmaf(crow[c], W[c * C_DIM + d], s);
    float* dst = (mat == 0) ? a : ((mat == 1) ? b : cg);
    dst[rem] = (mat < 2) ? tanhf(GCALPHA * s) : s;
}

// ---------------------------------------------------------------------------
// Kernel 1b (single workgroup): adjacency + top-k + sym-norm + cl = Anorm@cg,
// then pack the f16 B operands:
//   Bpack[g][col][k], g0/g1 = normalized refined codebook nc^T (cols 0-31),
//                     g2/g3 = raw clusters^T (cols 0-31), cols>=27 zero.
// ---------------------------------------------------------------------------
__global__ __launch_bounds__(256)
void refine_graph(const float* __restrict__ clusters,
                  const float* __restrict__ a, const float* __restrict__ b,
                  const float* __restrict__ cg,
                  float* __restrict__ cl_ws, float* __restrict__ cl_out,
                  _Float16* __restrict__ Bpack)
{
    __shared__ float adj[N_CLS * N_CLS];
    __shared__ float dinv[N_CLS];
    __shared__ float rn[N_CLS];
    const int tid = threadIdx.x;

    // adj = relu(tanh(3*(a@b^T - b@a^T)))
    for (int e = tid; e < N_CLS * N_CLS; e += 256) {
        int i = e / N_CLS, j = e % N_CLS;
        const float* ai = a + i * C_DIM; const float* bj = b + j * C_DIM;
        const float* bi = b + i * C_DIM; const float* aj = a + j * C_DIM;
        float d1 = 0.f, d2 = 0.f;
        for (int c = 0; c < C_DIM; ++c) {
            d1 = fmaf(ai[c], bj[c], d1);
            d2 = fmaf(bi[c], aj[c], d2);
        }
        float v = tanhf(GCALPHA * (d1 - d2));
        adj[e] = v > 0.f ? v : 0.f;
    }
    __syncthreads();

    // per row: 5th-largest threshold, sparsify in place, degree with self-loop
    if (tid < N_CLS) {
        float rowv[N_CLS];
        for (int j = 0; j < N_CLS; ++j) rowv[j] = adj[tid * N_CLS + j];
        float thr = 0.f;
        for (int k = 0; k < 5; ++k) {
            int bi_ = 0; float bv = -1.f;
            for (int j = 0; j < N_CLS; ++j)
                if (rowv[j] > bv) { bv = rowv[j]; bi_ = j; }
            thr = bv; rowv[bi_] = -1.f;
        }
        float s = 1.f; // self loop
        for (int j = 0; j < N_CLS; ++j) {
            float v = adj[tid * N_CLS + j];
            v = (v >= thr) ? v : 0.f;
            adj[tid * N_CLS + j] = v;   // keep sparsified values
            s += v;
        }
        dinv[tid] = 1.f / sqrtf(s);
    }
    __syncthreads();

    // cl = (dinv_i * Ahat_ij * dinv_j) @ cg
    for (int e = tid; e < N_CLS * C_DIM; e += 256) {
        int i = e / C_DIM, d = e % C_DIM;
        float s = 0.f;
        for (int j = 0; j < N_CLS; ++j) {
            float ahat = adj[i * N_CLS + j] + (i == j ? 1.f : 0.f);
            s = fmaf(ahat * dinv[j], cg[j * C_DIM + d], s);
        }
        float v = dinv[i] * s;
        cl_ws[e] = v;
        cl_out[e] = v;
    }
    __syncthreads();

    if (tid < N_CLS) {
        float s = 0.f;
        for (int d = 0; d < C_DIM; ++d) {
            float v = cl_ws[tid * C_DIM + d];
            s = fmaf(v, v, s);
        }
        rn[tid] = 1.f / fmaxf(sqrtf(s), 1e-12f);
    }
    __syncthreads();

    // pack B operands (f16, column-major per 16-wide group)
    for (int e = tid; e < 4 * 16 * C_DIM; e += 256) {
        int g = e / (16 * C_DIM);
        int rem = e % (16 * C_DIM);
        int col = rem / C_DIM;
        int k = rem % C_DIM;
        int n = (g & 1) * 16 + col;
        float v = 0.f;
        if (n < N_CLS)
            v = (g < 2) ? cl_ws[n * C_DIM + k] * rn[n] : clusters[n * C_DIM + k];
        Bpack[e] = (_Float16)v;
    }
}

// ---------------------------------------------------------------------------
// Kernel 2: fused dual GEMM + norms + softmax losses + transposed IP store.
// 256 threads = 8 waves; each wave owns 16 contiguous pixels; block = 128 px.
// ---------------------------------------------------------------------------
__global__ __launch_bounds__(256)
void main_fused(const float* __restrict__ x, const float* __restrict__ alphaP,
                const _Float16* __restrict__ Bpack,
                float* __restrict__ outIP,
                float* __restrict__ lossPart, float* __restrict__ entPart)
{
    __shared__ alignas(32) _Float16 ldsB[4 * 16 * C_DIM];   // 49152 B
    __shared__ float ldsIP[N_CLS * 132];                    // padded pitch
    __shared__ float red0[256];
    __shared__ float red1[256];

    const int tid = threadIdx.x;

    // stage shared B operand into LDS (3072 x float4)
    {
        const float4* s = (const float4*)Bpack;
        float4* d = (float4*)ldsB;
        for (int i = tid; i < 3072; i += 256) d[i] = s[i];
    }
    __syncthreads();

    const int lane = tid & 31;
    const int wave = tid >> 5;
    const int row  = lane & 15;    // pixel row (A) / N column (B,C)
    const int hi   = lane >> 4;

    const int p0   = blockIdx.x * 128 + wave * 16;
    const int bimg = p0 / HW_IMG;        // whole block in one image (6400%128==0)
    const int hw0  = p0 % HW_IMG;

    const float alpha = alphaP[0];
    const float* xbase = x + (size_t)bimg * C_DIM * HW_IMG + hw0 + row;

    v8f acc0 = {}; v8f acc1 = {};   // x . nc^T   (cols 0-15, 16-31)
    v8f acc2 = {}; v8f acc3 = {};   // x . clusters^T
    float nsq = 0.f;

    const int kloA  = hi * 16;  // A: lanes 0-15 -> K 0..15; 16-31 -> K 16..31
    const int koffB = hi * 16;  // B: lanes 0-15 -> K 0..15; 16-31 -> K 16..31

    for (int kc = 0; kc < C_DIM; kc += 32) {
        v16h af;
#pragma unroll
        for (int j = 0; j < 8; ++j) {
            float v = xbase[(size_t)(kc + kloA + j) * HW_IMG];
            nsq = fmaf(v, v, nsq);
            af[j] = (_Float16)v;
        }
#pragma unroll
        for (int j = 0; j < 8; ++j) {
            float v = xbase[(size_t)(kc + 8 + kloA + j) * HW_IMG];
            nsq = fmaf(v, v, nsq);
            af[8 + j] = (_Float16)v;
        }
        const int bb = kc + koffB;
        v16h bf0 = *(const v16h*)&ldsB[(0 * 16 + row) * C_DIM + bb];
        v16h bf1 = *(const v16h*)&ldsB[(1 * 16 + row) * C_DIM + bb];
        v16h bf2 = *(const v16h*)&ldsB[(2 * 16 + row) * C_DIM + bb];
        v16h bf3 = *(const v16h*)&ldsB[(3 * 16 + row) * C_DIM + bb];
        acc0 = __builtin_amdgcn_wmma_f32_16x16x32_f16(false, af, false, bf0, (short)0, acc0, false, false);
        acc1 = __builtin_amdgcn_wmma_f32_16x16x32_f16(false, af, false, bf1, (short)0, acc1, false, false);
        acc2 = __builtin_amdgcn_wmma_f32_16x16x32_f16(false, af, false, bf2, (short)0, acc2, false, false);
        acc3 = __builtin_amdgcn_wmma_f32_16x16x32_f16(false, af, false, bf3, (short)0, acc3, false, false);
    }

    // full ||x||^2 per pixel row: combine complementary K halves
    nsq += __shfl_xor(nsq, 16, 32);
    const float inv = 1.0f / fmaxf(sqrtf(nsq), 1e-12f);

    const bool v1 = (row < N_CLS - 16);  // acc1/acc3 column valid (n=16+row<27)
    float lossAcc = 0.f, entAcc = 0.f;

#pragma unroll
    for (int r = 0; r < 8; ++r) {
        const int   M    = r + hi * 8;                // pixel within wave tile
        const float invM = __shfl(inv, r + hi * 8, 32);

        float ip0 = acc0[r] * invM;
        float ip1 = acc1[r] * invM;

        // transposed deposit for coalesced store later
        ldsIP[row * 132 + wave * 16 + M] = ip0;
        if (v1) ldsIP[(16 + row) * 132 + wave * 16 + M] = ip1;

        // ---- cluster_loss: softmax_n(alpha*ip) ----
        float z0 = alpha * ip0;
        float z1 = alpha * ip1;
        float m = fmaxf(z0, v1 ? z1 : -1e30f);
#pragma unroll
        for (int k = 1; k < 16; k <<= 1) m = fmaxf(m, __shfl_xor(m, k, 32));
        float e0 = __expf(z0 - m);
        float e1 = v1 ? __expf(z1 - m) : 0.f;
        float S = e0 + e1;
        float T = e0 * ip0 + e1 * ip1;
#pragma unroll
        for (int k = 1; k < 16; k <<= 1) {
            S += __shfl_xor(S, k, 32);
            T += __shfl_xor(T, k, 32);
        }
        if (row == 0) lossAcc -= T / S;

        // ---- entropy: softmax_n(x . clusters) ----
        float s0 = acc2[r];
        float s1 = acc3[r];
        float m2 = fmaxf(s0, v1 ? s1 : -1e30f);
#pragma unroll
        for (int k = 1; k < 16; k <<= 1) m2 = fmaxf(m2, __shfl_xor(m2, k, 32));
        float f0 = __expf(s0 - m2);
        float f1 = v1 ? __expf(s1 - m2) : 0.f;
        float S2 = f0 + f1;
#pragma unroll
        for (int k = 1; k < 16; k <<= 1) S2 += __shfl_xor(S2, k, 32);
        float rS2 = 1.0f / S2;
        float p0p = f0 * rS2;
        float p1p = f1 * rS2;
        float t = p0p * __logf(p0p + 1e-10f) + (v1 ? p1p * __logf(p1p + 1e-10f) : 0.f);
#pragma unroll
        for (int k = 1; k < 16; k <<= 1) t += __shfl_xor(t, k, 32);
        if (row == 0) entAcc -= t;
    }

    __syncthreads();

    // coalesced inner_products store: [b, n, hw], 128 contiguous px per row
    {
        const int pb  = blockIdx.x * 128;
        const int bB  = pb / HW_IMG;
        const int hwB = pb % HW_IMG;
        float* ipOut = outIP + (size_t)bB * N_CLS * HW_IMG + hwB;
        for (int e = tid; e < N_CLS * 128; e += 256) {
            int n = e >> 7, px = e & 127;
            ipOut[(size_t)n * HW_IMG + px] = ldsIP[n * 132 + px];
        }
    }

    // deterministic block reduction of loss/entropy partials
    red0[tid] = lossAcc;
    red1[tid] = entAcc;
    __syncthreads();
#pragma unroll
    for (int s = 128; s > 0; s >>= 1) {
        if (tid < s) { red0[tid] += red0[tid + s]; red1[tid] += red1[tid + s]; }
        __syncthreads();
    }
    if (tid == 0) {
        lossPart[blockIdx.x] = red0[0];
        entPart[blockIdx.x]  = red1[0];
    }
}

// ---------------------------------------------------------------------------
// Kernel 3: deterministic fixed-order final reduction + mean.
// ---------------------------------------------------------------------------
__global__ __launch_bounds__(256)
void finalize_k(const float* __restrict__ lossPart, const float* __restrict__ entPart,
                float* __restrict__ outLoss, float* __restrict__ outEnt)
{
    __shared__ float r0[256];
    __shared__ float r1[256];
    const int tid = threadIdx.x;
    float s0 = 0.f, s1 = 0.f;
    for (int i = tid; i < NBLK; i += 256) { s0 += lossPart[i]; s1 += entPart[i]; }
    r0[tid] = s0; r1[tid] = s1;
    __syncthreads();
#pragma unroll
    for (int s = 128; s > 0; s >>= 1) {
        if (tid < s) { r0[tid] += r0[tid + s]; r1[tid] += r1[tid + s]; }
        __syncthreads();
    }
    if (tid == 0) {
        *outLoss = r0[0] / (float)N_PIX;
        *outEnt  = r1[0] / (float)N_PIX;
    }
}

// ---------------------------------------------------------------------------
extern "C" void kernel_launch(void* const* d_in, const int* in_sizes, int n_in,
                              void* d_out, int out_size, void* d_ws, size_t ws_size,
                              hipStream_t stream)
{
    (void)in_sizes; (void)n_in; (void)out_size; (void)ws_size;

    const float* x        = (const float*)d_in[0];
    const float* alphaP   = (const float*)d_in[1];
    const float* clusters = (const float*)d_in[2];
    const float* gc_w1    = (const float*)d_in[3];
    const float* gc_w2    = (const float*)d_in[4];
    const float* gcn_w    = (const float*)d_in[5];

    // output tuple flat: loss(1) | inner_products(B*N*H*W) | cl(N*C) | entropy(1)
    float* out     = (float*)d_out;
    float* outLoss = out;
    float* outIP   = out + 1;
    float* outCl   = outIP + (size_t)B_IMG * N_CLS * HW_IMG;
    float* outEnt  = outCl + (size_t)N_CLS * C_DIM;

    // workspace layout (16B-aligned sections)
    char* w = (char*)d_ws;
    float*    ws_a  = (float*)w;    w += (size_t)N_CLS * C_DIM * sizeof(float);
    float*    ws_b  = (float*)w;    w += (size_t)N_CLS * C_DIM * sizeof(float);
    float*    ws_cg = (float*)w;    w += (size_t)N_CLS * C_DIM * sizeof(float);
    float*    ws_cl = (float*)w;    w += (size_t)N_CLS * C_DIM * sizeof(float);
    _Float16* ws_Bp = (_Float16*)w; w += (size_t)4 * 16 * C_DIM * sizeof(_Float16);
    float*    ws_lp = (float*)w;    w += (size_t)NBLK * sizeof(float);
    float*    ws_ep = (float*)w;    w += (size_t)NBLK * sizeof(float);

    refine_mm<<<(3 * N_CLS * C_DIM + 255) / 256, 256, 0, stream>>>(
        clusters, gc_w1, gc_w2, gcn_w, ws_a, ws_b, ws_cg);
    refine_graph<<<1, 256, 0, stream>>>(
        clusters, ws_a, ws_b, ws_cg, ws_cl, outCl, ws_Bp);
    main_fused<<<NBLK, 256, 0, stream>>>(
        x, alphaP, ws_Bp, outIP, ws_lp, ws_ep);
    finalize_k<<<1, 256, 0, stream>>>(ws_lp, ws_ep, outLoss, outEnt);
}
